// SNNClassifier_32667521253571
// MI455X (gfx1250) — matrix operand order repaired and checked by reading, not compiled, not run on hardware
//
#include <hip/hip_runtime.h>
#include <hip/hip_bf16.h>

// MI455X / gfx1250, wave32. Matrix ops via V_WMMA_F32_16X16X32_BF16.
typedef __attribute__((ext_vector_type(16))) __bf16 v16bf;
typedef __attribute__((ext_vector_type(8)))  float  v8f;

#define SNN_B 32
#define SNN_T 1024
#define SNN_D 256
#define SNN_H 512
#define SNN_C 10
#define BETA 0.9f

// ---------------------------------------------------------------------------
// Pack an fp32 KxN (N=512) weight matrix into bf16 "B-matrix" tile layout:
// 32x16 tiles, per tile: element (k,n) -> lane = ((k>>3)&1)*16 + (n&15),
// e = ((k>>4)&1)*8 + (k&7); tile block = (kt*32 + nt)*512 elements.
// ---------------------------------------------------------------------------
__global__ __launch_bounds__(256) void snn_pack_b(const float* __restrict__ src,
                                                  void* __restrict__ dstv,
                                                  int total) {
    __bf16* dst = (__bf16*)dstv;
    int idx = blockIdx.x * 256 + threadIdx.x;
    if (idx >= total) return;
    int k = idx >> 9;          // N = 512
    int n = idx & 511;
    int off = ((k >> 5) * 32 + (n >> 4)) * 512
            + ((((k >> 3) & 1) * 16 + (n & 15)) * 16)
            + (((k >> 4) & 1) * 8 + (k & 7));
    dst[off] = (__bf16)src[idx];
}

// ---------------------------------------------------------------------------
// Phase 1: cur[t][b][h] = x[b][t][:] @ W_in + b_in   (bf16 WMMA, f32 accum)
// One wave computes a 16x64 output tile. Grid: 2048 m-tiles x 8 n-quads.
// ---------------------------------------------------------------------------
__global__ __launch_bounds__(256) void snn_ingemm(const float* __restrict__ x,
                                                  const void* __restrict__ WinPv,
                                                  const float* __restrict__ b_in,
                                                  float* __restrict__ cur) {
    const __bf16* WinP = (const __bf16*)WinPv;
    const int lane  = threadIdx.x & 31;
    const int wv    = threadIdx.x >> 5;
    const int gw    = blockIdx.x * 8 + wv;   // 16384 waves total
    const int mtile = gw >> 3;               // 0..2047 (B*T/16)
    const int nquad = gw & 7;                // 0..7   (H/64)
    const int laneN = lane & 15;
    const int laneH = lane >> 4;

    const int row = mtile * 16 + laneN;      // global row = b*T + t
    const float* xrow = x + row * SNN_D;

    v8f acc[4];
#pragma unroll
    for (int q = 0; q < 4; ++q) {
        float bv = b_in[nquad * 64 + q * 16 + laneN];
#pragma unroll
        for (int r = 0; r < 8; ++r) acc[q][r] = bv;
    }

#pragma unroll
    for (int kt = 0; kt < SNN_D / 32; ++kt) {
        // A tile (16x32) per ISA layout: lane holds row m=laneN,
        // pair j covers k' = (j>>2)*16 + laneH*8 + (j&3)*2 + {0,1}
        v16bf a;
#pragma unroll
        for (int j = 0; j < 8; ++j) {
            int k0 = kt * 32 + (j >> 2) * 16 + laneH * 8 + (j & 3) * 2;
            float2 f = *(const float2*)(xrow + k0);
            a[2 * j]     = (__bf16)f.x;
            a[2 * j + 1] = (__bf16)f.y;
        }
#pragma unroll
        for (int q = 0; q < 4; ++q) {
            v16bf b = *(const v16bf*)(WinP + (kt * 32 + nquad * 4 + q) * 512 + lane * 16);
            acc[q] = __builtin_amdgcn_wmma_f32_16x16x32_bf16(
                false, a, false, b, (short)0, acc[q], false, false);
        }
    }

    // store to cur[(t*32 + b)*512 + n]  (scan-friendly [T][B][H] order)
#pragma unroll
    for (int q = 0; q < 4; ++q) {
        const int n = nquad * 64 + q * 16 + laneN;
#pragma unroll
        for (int r = 0; r < 8; ++r) {
            int m    = laneH * 8 + r;
            int grow = mtile * 16 + m;          // = b*T + t
            int bb   = grow >> 10;
            int tt   = grow & 1023;
            cur[((tt * SNN_B + bb) << 9) + n] = acc[q][r];
        }
    }
}

// ---------------------------------------------------------------------------
// Phase 2: persistent single-WGP recurrent LIF scan + fused readout.
// 512 threads = 16 waves; wave w owns output columns [32w, 32w+32):
//   - W_rec bf16 B-tiles resident in VGPRs (16 kt x 2 n-tiles = 256 VGPRs)
//   - spk double-buffered in LDS in packed A-matrix layout (2 x 32KB)
//   - v kept in WMMA accumulators for all 1024 steps
//   - cur_t reads software-pipelined in a parity-toggled register buffer so
//     the per-step L2 latency hides behind the WMMA chain of the prior step
// ---------------------------------------------------------------------------
__global__ __launch_bounds__(512) void snn_scan(const float* __restrict__ cur,
                                                const void* __restrict__ WrecPv,
                                                const float* __restrict__ W_ro,
                                                const float* __restrict__ b_ro,
                                                float* __restrict__ out) {
    __shared__ __align__(16) unsigned char smem[65536];  // spk[2][16384] bf16 / vbuf alias
    __shared__ unsigned int cnt_sh;
    __bf16* spk  = (__bf16*)smem;
    float*  vbuf = (float*)smem;   // reused after the scan
    const __bf16* WrecP = (const __bf16*)WrecPv;

    const int tid   = threadIdx.x;
    const int w     = tid >> 5;    // wave 0..15
    const int lane  = tid & 31;
    const int laneN = lane & 15;
    const int laneH = lane >> 4;

    // zero spike buffer 0 (state at t=0) and the counter
    {
        unsigned int* z = (unsigned int*)smem;
        for (int i = tid; i < 8192; i += 512) z[i] = 0u;   // 16384 bf16
        if (tid == 0) cnt_sh = 0u;
    }
    __syncthreads();

    // W_rec resident in registers: n-tiles 2w, 2w+1, all 16 k-tiles
    v16bf Brec[16][2];
#pragma unroll
    for (int kt = 0; kt < 16; ++kt)
#pragma unroll
        for (int q = 0; q < 2; ++q)
            Brec[kt][q] = *(const v16bf*)(WrecP + (kt * 32 + (2 * w + q)) * 512 + lane * 16);

    v8f acc[2][2] = {};            // v state: [m-tile][n-tile]
    unsigned int cnt = 0;

    // prologue: load cur for t=0 into parity buffer 0
    float curR[2][2][2][8];        // [parity][mi][q][r]
#pragma unroll
    for (int mi = 0; mi < 2; ++mi)
#pragma unroll
        for (int q = 0; q < 2; ++q) {
            const int n = (2 * w + q) * 16 + laneN;
#pragma unroll
            for (int r = 0; r < 8; ++r) {
                const int m = mi * 16 + laneH * 8 + r;
                curR[0][mi][q][r] = cur[(m << 9) + n];
            }
        }

    for (int t = 0; t < SNN_T; ++t) {
        const int pb = t & 1;                    // parity: consume buffer
        const int nb = pb ^ 1;                   // parity: fill buffer
        const int ib = pb * 16384;               // input spike buffer (elements)
        const int ob = nb * 16384;               // output spike buffer
        const int tn = (t + 1 < SNN_T) ? t + 1 : t;   // branchless next-step index

        // v = beta*v + cur_t (from pipelined registers)
#pragma unroll
        for (int mi = 0; mi < 2; ++mi)
#pragma unroll
            for (int q = 0; q < 2; ++q)
#pragma unroll
                for (int r = 0; r < 8; ++r)
                    acc[mi][q][r] = BETA * acc[mi][q][r] + curR[pb][mi][q][r];

        // v += spk @ W_rec  (spikes already packed in A layout in LDS)
#pragma unroll
        for (int kt = 0; kt < 16; ++kt) {
            v16bf a0 = *(const v16bf*)(spk + ib + (kt) * 512 + lane * 16);
            v16bf a1 = *(const v16bf*)(spk + ib + (16 + kt) * 512 + lane * 16);
#pragma unroll
            for (int q = 0; q < 2; ++q) {
                acc[0][q] = __builtin_amdgcn_wmma_f32_16x16x32_bf16(
                    false, a0, false, Brec[kt][q], (short)0, acc[0][q], false, false);
                acc[1][q] = __builtin_amdgcn_wmma_f32_16x16x32_bf16(
                    false, a1, false, Brec[kt][q], (short)0, acc[1][q], false, false);
            }
        }

        // issue next-step cur loads now: they overlap the in-flight WMMA chain,
        // the threshold pass below, and the barrier
#pragma unroll
        for (int mi = 0; mi < 2; ++mi)
#pragma unroll
            for (int q = 0; q < 2; ++q) {
                const int n = (2 * w + q) * 16 + laneN;
#pragma unroll
                for (int r = 0; r < 8; ++r) {
                    const int m = mi * 16 + laneH * 8 + r;
                    curR[nb][mi][q][r] = cur[((tn * SNN_B + m) << 9) + n];
                }
            }
        // pull the t+2 slab (64KB) into the WGP cache: 512 threads x 128B stride
        {
            const int t2 = (t + 2 < SNN_T) ? t + 2 : t;
            __builtin_prefetch(cur + ((size_t)t2 * SNN_B * SNN_H) + tid * 32, 0, 1);
        }

        // threshold, subtractive reset, count, emit spikes for step t+1
#pragma unroll
        for (int mi = 0; mi < 2; ++mi)
#pragma unroll
            for (int q = 0; q < 2; ++q) {
                const int kg  = (2 * w + q) * 16 + laneN;    // next-step K index
                const int kt2 = kg >> 5;
                const int lh  = (kg >> 3) & 1;
                const int e   = ((kg >> 4) & 1) * 8 + ((kg >> 1) & 3) * 2 + (kg & 1);
#pragma unroll
                for (int r = 0; r < 8; ++r) {
                    const int m = mi * 16 + laneH * 8 + r;
                    float v  = acc[mi][q][r];
                    bool  s  = v >= 1.0f;
                    acc[mi][q][r] = s ? v - 1.0f : v;
                    cnt += s ? 1u : 0u;
                    spk[ob + (mi * 16 + kt2) * 512 + (lh * 16 + (m & 15)) * 16 + e] =
                        s ? (__bf16)1.0f : (__bf16)0.0f;
                }
            }
        __syncthreads();
    }

    // dump v to LDS (spike buffers are dead), then readout + sparsity
#pragma unroll
    for (int mi = 0; mi < 2; ++mi)
#pragma unroll
        for (int q = 0; q < 2; ++q) {
            const int n = (2 * w + q) * 16 + laneN;
#pragma unroll
            for (int r = 0; r < 8; ++r) {
                const int m = mi * 16 + laneH * 8 + r;
                vbuf[(m << 9) + n] = acc[mi][q][r];
            }
        }
    atomicAdd(&cnt_sh, cnt);
    __syncthreads();

    for (int i = tid; i < SNN_B * SNN_C; i += 512) {
        int b = i / SNN_C, c = i % SNN_C;
        float s = b_ro[c];
        for (int k = 0; k < SNN_H; ++k)
            s += vbuf[(b << 9) + k] * W_ro[k * SNN_C + c];
        out[i] = s;
    }
    if (tid == 0)
        out[SNN_B * SNN_C] =
            1.0f - (float)cnt_sh / (float)(SNN_B * SNN_T * SNN_H);
}

// ---------------------------------------------------------------------------
extern "C" void kernel_launch(void* const* d_in, const int* in_sizes, int n_in,
                              void* d_out, int out_size, void* d_ws, size_t ws_size,
                              hipStream_t stream) {
    (void)in_sizes; (void)n_in; (void)out_size; (void)ws_size;
    const float* x     = (const float*)d_in[0];
    const float* W_in  = (const float*)d_in[1];
    const float* b_in  = (const float*)d_in[2];
    const float* W_rec = (const float*)d_in[3];
    const float* W_ro  = (const float*)d_in[4];
    const float* b_ro  = (const float*)d_in[5];
    float* out = (float*)d_out;

    char* ws = (char*)d_ws;
    float* cur  = (float*)ws;                              // T*B*H f32 = 64 MiB
    void*  WrecP = (void*)(ws + 67108864);                 // 512x512 bf16 = 512 KiB
    void*  WinP  = (void*)(ws + 67108864 + 524288);        // 256x512 bf16 = 256 KiB

    snn_pack_b<<<(SNN_D * SNN_H + 255) / 256, 256, 0, stream>>>(W_in,  WinP,  SNN_D * SNN_H);
    snn_pack_b<<<(SNN_H * SNN_H + 255) / 256, 256, 0, stream>>>(W_rec, WrecP, SNN_H * SNN_H);
    snn_ingemm<<<2048, 256, 0, stream>>>(x, WinP, b_in, cur);
    snn_scan<<<1, 512, 0, stream>>>(cur, WrecP, W_ro, b_ro, out);
}
